// Encoder_37477884625046
// MI455X (gfx1250) — matrix operand order repaired and checked by reading, not compiled, not run on hardware
//
#include <hip/hip_runtime.h>
#include <math.h>
#include <stdint.h>

#define TT    512
#define BB    64
#define II    512
#define CELLD 512
#define HHD   1024
#define GG    2048   // 4*CELL
#define KD    1024   // I + CELL
#define NBLK  32     // persistent LSTM grid size
#define APAD  1032   // 1024 + 8 halfs: conflict-free LDS row stride

typedef _Float16 h16;
typedef __attribute__((ext_vector_type(16))) _Float16 h16x16;
typedef __attribute__((ext_vector_type(8)))  _Float16 h16x8;
typedef __attribute__((ext_vector_type(8)))  float    f32x8;

// ---------------------------------------------------------------- helpers
__device__ __forceinline__ f32x8 wmma16(h16x16 a, h16x16 b, f32x8 c) {
  // D = A(16x32 f16) * B(32x16 f16) + C(16x16 f32)
  return __builtin_amdgcn_wmma_f32_16x16x32_f16(false, a, false, b,
                                                (short)0, c, false, false);
}
__device__ __forceinline__ h16x16 frag_halves(const h16* lo, const h16* hi) {
  h16x8 l = *(const h16x8*)lo;
  h16x8 h = *(const h16x8*)hi;
  return __builtin_shufflevector(l, h, 0,1,2,3,4,5,6,7,8,9,10,11,12,13,14,15);
}
__device__ __forceinline__ h16x16 frag_contig(const h16* p) {
  return *(const h16x16*)p;
}
__device__ __forceinline__ f32x8 zero8() {
  f32x8 v;
#pragma unroll
  for (int i = 0; i < 8; ++i) v[i] = 0.f;
  return v;
}
__device__ __forceinline__ float sigf(float x) { return 1.f / (1.f + __expf(-x)); }

// ---------------------------------------------------------------- prep kernels
__global__ __launch_bounds__(256) void prep_weights(
    const float* __restrict__ WihF, const float* __restrict__ WhhF,
    const float* __restrict__ WihB, const float* __restrict__ WhhB,
    h16* __restrict__ WfH, h16* __restrict__ WbH) {
  size_t idx = (size_t)blockIdx.x * 256 + threadIdx.x;       // < 2*GG*KD
  int    dir = (int)(idx / ((size_t)GG * KD));
  size_t rem = idx % ((size_t)GG * KD);
  int r = (int)(rem / KD), k = (int)(rem % KD);
  const float* Wih = dir ? WihB : WihF;
  const float* Whh = dir ? WhhB : WhhF;
  float v = (k < II) ? Wih[(size_t)r * II + k] : Whh[(size_t)r * CELLD + (k - II)];
  (dir ? WbH : WfH)[rem] = (h16)v;
}

__global__ __launch_bounds__(256) void prep_f32_to_f16(
    const float* __restrict__ src, h16* __restrict__ dst) {
  size_t i = (size_t)blockIdx.x * 256 + threadIdx.x;
  dst[i] = (h16)src[i];
}

__global__ __launch_bounds__(256) void init_misc(
    const float* __restrict__ bihF, const float* __restrict__ bhhF,
    const float* __restrict__ bihB, const float* __restrict__ bhhB,
    float* biasF, float* biasB,
    float* hF, float* cF, float* hB, float* cB,
    h16* hxF0, h16* hxF1, h16* hxB0, h16* hxB1, float* wsum,
    unsigned* bar) {
  int i = blockIdx.x * 256 + threadIdx.x;                    // < 32768
  hF[i] = cF[i] = hB[i] = cB[i] = 0.f;
  hxF0[i] = hxF1[i] = hxB0[i] = hxB1[i] = (h16)0.f;
  wsum[i] = 0.f;                                             // BB*TT == 32768
  if (i < GG) { biasF[i] = bihF[i] + bhhF[i]; biasB[i] = bihB[i] + bhhB[i]; }
  if (i == 0) *bar = 0u;
}

// ---------------------------------------------------------------- LSTM (persistent)
// Single launch runs all 512 timesteps. 32 co-resident workgroups:
// blockIdx [0,16)=fwd, [16,32)=bwd, 32 cells (all 4 gates) per block.
// Per wave: whole 16-col weight slice in 256 VGPRs (32 B-fragments),
// double-buffered A-fragment pipeline, h/c state register-resident,
// device-wide monotonic atomic barrier between steps.
__global__ __launch_bounds__(256)
__attribute__((amdgpu_waves_per_eu(2)))
void lstm_persistent(
    const h16* __restrict__ xh, const h16* __restrict__ Wf, const h16* __restrict__ Wb,
    const float* __restrict__ biasF, const float* __restrict__ biasB,
    const int* __restrict__ length,
    float* hF, float* cF, float* hB, float* cB,
    h16* hxF0, h16* hxF1, h16* hxB0, h16* hxB1,
    float* __restrict__ outF, h16* __restrict__ outH,
    unsigned* bar) {
  __shared__ float gb[BB][128];                              // 32 KB gate buffer
  __shared__ float sbias[128];
  __shared__ int   slen[BB];

  const int tid  = threadIdx.x;
  const int wave = tid >> 5, lane = tid & 31;
  const int half = lane >> 4, lm = lane & 15;
  const int dir  = blockIdx.x >> 4;
  const int c0   = (blockIdx.x & 15) * 32;

  const h16*   W    = dir ? Wb : Wf;
  const float* bias = dir ? biasB : biasF;
  float* hS = dir ? hB : hF;
  float* cS = dir ? cB : cF;
  h16* hxA   = dir ? hxB0 : hxF0;
  h16* hxBuf = dir ? hxB1 : hxF1;

  // wave -> one N-tile: gate = wave>>1, cell-subtile = wave&1
  const int gate = wave >> 1, csub = wave & 1;
  const int gcol = gate * CELLD + c0 + csub * 16;
  const h16* wrow = W + (size_t)(gcol + lm) * KD;            // B lane: col N=lm

  if (tid < 128) {
    int g = tid >> 5, c = tid & 31;
    sbias[tid] = bias[g * CELLD + c0 + c];
  }
  if (tid < BB) slen[tid] = length[tid];

  // register-resident weight slice: 32 k-steps x 8 VGPRs = 256 VGPRs
  h16x16 wf[32];
#pragma unroll
  for (int ks = 0; ks < 32; ++ks)
    wf[ks] = frag_halves(wrow + ks * 32 + half * 8, wrow + ks * 32 + 16 + half * 8);
  __syncthreads();

  // register-resident cell state: thread owns 8 fixed (b, cell) pairs
  float hreg[8], creg[8];
#pragma unroll
  for (int i = 0; i < 8; ++i) { hreg[i] = 0.f; creg[i] = 0.f; }

  for (int s = 0; s < TT; ++s) {
    const int t = dir ? (TT - 1 - s) : s;
    const h16* hxin  = (s & 1) ? hxBuf : hxA;
    h16*       hxout = (s & 1) ? hxA : hxBuf;
    const h16* xpanel = xh + (size_t)t * BB * II;

    // prefetch next step's x panel (64 KB) while this step computes
    {
      int tn = dir ? (TT - 2 - s) : (s + 1);
      if (tn >= 0 && tn < TT) {
        const char* p = (const char*)(xh + (size_t)tn * BB * II);
        __builtin_prefetch(p + tid * 256, 0, 0);
        __builtin_prefetch(p + tid * 256 + 128, 0, 0);
      }
    }

    auto load_af = [&](h16x16* dst, int kk) {
#pragma unroll
      for (int m = 0; m < 4; ++m) {
        int row = m * 16 + lm;                               // A lane: row M=lm
        const h16* arow = (kk < 16)
            ? (xpanel + (size_t)row * II + kk * 32)
            : (hxin + (size_t)row * CELLD + (kk - 16) * 32);
        dst[m] = frag_halves(arow + half * 8, arow + 16 + half * 8);
      }
    };

    f32x8 acc[4];
#pragma unroll
    for (int m = 0; m < 4; ++m) acc[m] = zero8();

    // software-pipelined K loop: load k-step n+1 while multiplying k-step n
    h16x16 afA[4], afB[4];
    load_af(afA, 0);
#pragma unroll
    for (int ks = 0; ks < 32; ks += 2) {
      load_af(afB, ks + 1);
#pragma unroll
      for (int m = 0; m < 4; ++m) acc[m] = wmma16(afA[m], wf[ks], acc[m]);
      if (ks + 2 < 32) load_af(afA, ks + 2);
#pragma unroll
      for (int m = 0; m < 4; ++m) acc[m] = wmma16(afB[m], wf[ks + 1], acc[m]);
    }

    // C layout: row = mt*16 + 8*half + r ; local col = wave*16 + lm
#pragma unroll
    for (int m = 0; m < 4; ++m)
#pragma unroll
      for (int r = 0; r < 8; ++r)
        gb[m * 16 + half * 8 + r][wave * 16 + lm] = acc[m][r];
    __syncthreads();

    // pointwise LSTM cell with pack_padded masking (state in registers)
#pragma unroll
    for (int it = 0; it < 8; ++it) {
      int idx = tid + it * 256;                              // < 2048
      int b = idx >> 5, c = idx & 31;
      int cg = c0 + c;
      float gi = gb[b][0 * 32 + c] + sbias[0 * 32 + c];
      float gf = gb[b][1 * 32 + c] + sbias[1 * 32 + c];
      float gg = gb[b][2 * 32 + c] + sbias[2 * 32 + c];
      float go = gb[b][3 * 32 + c] + sbias[3 * 32 + c];
      float iv = sigf(gi), fv = sigf(gf), gv = tanhf(gg), ov = sigf(go);
      float cn = fv * creg[it] + iv * gv;
      float hn = ov * tanhf(cn);
      bool  msk = t < slen[b];
      float c2 = msk ? cn : creg[it];
      float h2 = msk ? hn : hreg[it];
      float ov2 = msk ? hn : 0.f;
      creg[it] = c2; hreg[it] = h2;
      hxout[(size_t)b * CELLD + cg] = (h16)h2;
      size_t oidx = ((size_t)t * BB + b) * HHD + dir * CELLD + cg;
      outF[oidx] = ov2;
      outH[oidx] = (h16)ov2;
    }
    __syncthreads();

    // device-wide step barrier (monotonic counter, agent scope)
    if (lane == 0) {
      if (tid == 0) {
        __threadfence();
        atomicAdd(bar, 1u);
      }
      unsigned target = (unsigned)NBLK * (unsigned)(s + 1);
      while (__hip_atomic_load(bar, __ATOMIC_ACQUIRE, __HIP_MEMORY_SCOPE_AGENT) < target)
        __builtin_amdgcn_s_sleep(1);
    }
    __syncthreads();
  }

  // dump final register state for finalize()
#pragma unroll
  for (int it = 0; it < 8; ++it) {
    int idx = tid + it * 256;
    int b = idx >> 5, c = idx & 31;
    size_t sidx = (size_t)b * CELLD + c0 + c;
    hS[sidx] = hreg[it];
    cS[sidx] = creg[it];
  }
}

// ---------------------------------------------------------------- attention
// Block = (t-slab of 16 rows, one batch b). A-slab (16x1024 f16) staged into
// LDS once via async global->LDS DMA, then scores = O_b * O_b^T via WMMA
// (A from LDS, B streamed), softmax over s in LDS (aliased with A stage),
// column-sums accumulated into wsum[b,s].
__global__ __launch_bounds__(256) void attn_kernel(
    const h16* __restrict__ outH, float* __restrict__ wsum) {
  // union: A stage = 16 x APAD halfs (33024 B) ; scores = 16 x 512 f32 (32768 B)
  __shared__ __align__(16) unsigned char smem[16 * APAD * 2];
  h16*  astage = (h16*)smem;
  float (*sc)[TT] = (float (*)[TT])smem;

  const int tid = threadIdx.x, wave = tid >> 5, lane = tid & 31;
  const int half = lane >> 4, lm = lane & 15;
  const int b  = blockIdx.y;
  const int t0 = blockIdx.x * 16;

  // async-stage A slab: 16 rows x 1024 halfs, 16B chunks, padded rows
#pragma unroll
  for (int i = 0; i < 8; ++i) {
    int q   = tid + i * 256;                                 // < 2048 chunks
    int row = q >> 7;                                        // 128 chunks/row
    int col = (q & 127) * 8;
    unsigned ldsa = (unsigned)(uintptr_t)(astage + row * APAD + col);
    uint64_t ga = (uint64_t)(uintptr_t)(outH +
        ((size_t)(t0 + row) * BB + b) * HHD + col);
    asm volatile("global_load_async_to_lds_b128 %0, %1, off"
                 :: "v"(ldsa), "v"(ga) : "memory");
  }
  asm volatile("s_wait_asynccnt 0x0" ::: "memory");
  __syncthreads();

  const h16* abase = astage + lm * APAD;                     // A lane: row M=lm
  f32x8 acc[4];
#pragma unroll
  for (int j = 0; j < 4; ++j) acc[j] = zero8();

  for (int k0 = 0; k0 < HHD; k0 += 32) {
    h16x16 afrag = frag_halves(abase + k0 + half * 8, abase + k0 + 16 + half * 8);
#pragma unroll
    for (int j = 0; j < 4; ++j) {
      int n0 = (wave + j * 8) * 16;
      const h16* brow = outH + ((size_t)(n0 + lm) * BB + b) * HHD;
      h16x16 bfrag = frag_halves(brow + k0 + half * 8, brow + k0 + 16 + half * 8);
      acc[j] = wmma16(afrag, bfrag, acc[j]);
    }
  }
  __syncthreads();                                           // astage dead; reuse as sc
#pragma unroll
  for (int j = 0; j < 4; ++j) {
    int n0 = (wave + j * 8) * 16;
#pragma unroll
    for (int r = 0; r < 8; ++r) sc[half * 8 + r][n0 + lm] = acc[j][r];
  }
  __syncthreads();

  // softmax: wave handles rows 2w, 2w+1
  for (int rr = 0; rr < 2; ++rr) {
    int r = wave * 2 + rr;
    float mx = -1e30f;
    for (int c = lane; c < TT; c += 32) mx = fmaxf(mx, sc[r][c]);
    for (int off = 16; off; off >>= 1) mx = fmaxf(mx, __shfl_xor(mx, off));
    float sum = 0.f;
    for (int c = lane; c < TT; c += 32) {
      float e = __expf(sc[r][c] - mx);
      sc[r][c] = e; sum += e;
    }
    for (int off = 16; off; off >>= 1) sum += __shfl_xor(sum, off);
    float inv = 1.f / sum;
    for (int c = lane; c < TT; c += 32) sc[r][c] *= inv;
  }
  __syncthreads();

  for (int c = tid; c < TT; c += 256) {
    float s = 0.f;
#pragma unroll
    for (int r = 0; r < 16; ++r) s += sc[r][c];
    atomicAdd(&wsum[(size_t)b * TT + c], s);
  }
}

// ---------------------------------------------------------------- context
__global__ __launch_bounds__(256) void ctx_gemv(
    const float* __restrict__ wsum, const float* __restrict__ outF,
    float* __restrict__ ctxin, h16* __restrict__ ctxinH) {
  int b = blockIdx.y;
  int h = blockIdx.x * 256 + threadIdx.x;
  const float* w = wsum + (size_t)b * TT;
  float acc = 0.f;
  for (int s2 = 0; s2 < TT; ++s2)
    acc += w[s2] * outF[((size_t)s2 * BB + b) * HHD + h];
  ctxin[(size_t)b * HHD + h]  = acc;
  ctxinH[(size_t)b * HHD + h] = (h16)acc;
}

__global__ __launch_bounds__(256) void ctx_proj(
    const h16* __restrict__ ctxinH, const h16* __restrict__ WcH,
    const float* __restrict__ bctx, float* __restrict__ outCtx) {
  int tid = threadIdx.x, wave = tid >> 5, lane = tid & 31;
  int half = lane >> 4, lm = lane & 15;
  int n0 = blockIdx.x * 128 + wave * 16;
  const h16* wrow = WcH + (size_t)(n0 + lm) * HHD;
  f32x8 acc[4];
#pragma unroll
  for (int m = 0; m < 4; ++m) acc[m] = zero8();
  for (int k0 = 0; k0 < HHD; k0 += 32) {
    h16x16 bfrag = frag_halves(wrow + k0 + half * 8, wrow + k0 + 16 + half * 8);
#pragma unroll
    for (int m = 0; m < 4; ++m) {
      const h16* arow = ctxinH + (size_t)(m * 16 + lm) * HHD + k0;
      h16x16 afrag = frag_halves(arow + half * 8, arow + 16 + half * 8);
      acc[m] = wmma16(afrag, bfrag, acc[m]);
    }
  }
  float bv = bctx[n0 + lm];
#pragma unroll
  for (int m = 0; m < 4; ++m)
#pragma unroll
    for (int r = 0; r < 8; ++r) {
      int row = m * 16 + half * 8 + r;
      outCtx[(size_t)row * HHD + n0 + lm] = acc[m][r] + bv;
    }
}

__global__ __launch_bounds__(256) void finalize(
    const float* __restrict__ hF, const float* __restrict__ hB,
    const float* __restrict__ cF, const float* __restrict__ cB,
    float* __restrict__ dout, size_t offH, size_t offC) {
  int idx = blockIdx.x * 256 + threadIdx.x;                  // < 2*BB*HHD
  int which = idx >> 16;                                     // BB*HHD == 65536
  int rem = idx & 65535;
  int b = rem >> 10, col = rem & 1023;
  const float* src = which ? (col < CELLD ? cF : cB) : (col < CELLD ? hF : hB);
  float v = src[b * CELLD + (col & (CELLD - 1))];
  dout[(which ? offC : offH) + rem] = v;
}

// ---------------------------------------------------------------- launcher
extern "C" void kernel_launch(void* const* d_in, const int* in_sizes, int n_in,
                              void* d_out, int out_size, void* d_ws, size_t ws_size,
                              hipStream_t stream) {
  (void)in_sizes; (void)n_in; (void)out_size; (void)ws_size;
  const float* x     = (const float*)d_in[0];
  const int*   len   = (const int*)  d_in[1];
  const float* WihF  = (const float*)d_in[2];
  const float* WhhF  = (const float*)d_in[3];
  const float* bihF  = (const float*)d_in[4];
  const float* bhhF  = (const float*)d_in[5];
  const float* WihB  = (const float*)d_in[6];
  const float* WhhB  = (const float*)d_in[7];
  const float* bihB  = (const float*)d_in[8];
  const float* bhhB  = (const float*)d_in[9];
  const float* Wctx  = (const float*)d_in[10];
  const float* bctx  = (const float*)d_in[11];
  float* dout = (float*)d_out;

  const size_t OUT_H   = (size_t)TT * BB * HHD;
  const size_t OUT_C   = OUT_H + (size_t)BB * HHD;
  const size_t OUT_CTX = OUT_C + (size_t)BB * HHD;

  char* wsb = (char*)d_ws;
  size_t off = 0;
  auto take = [&](size_t bytes) -> void* {
    void* p = wsb + off;
    off = (off + bytes + 255) & ~(size_t)255;
    return p;
  };
  h16*   WfH    = (h16*)  take((size_t)GG * KD * 2);
  h16*   WbH    = (h16*)  take((size_t)GG * KD * 2);
  h16*   WcH    = (h16*)  take((size_t)HHD * HHD * 2);
  float* biasF  = (float*)take((size_t)GG * 4);
  float* biasB  = (float*)take((size_t)GG * 4);
  h16*   xH     = (h16*)  take((size_t)TT * BB * II * 2);
  h16*   outH   = (h16*)  take((size_t)TT * BB * HHD * 2);
  float* hF     = (float*)take((size_t)BB * CELLD * 4);
  float* cF     = (float*)take((size_t)BB * CELLD * 4);
  float* hB     = (float*)take((size_t)BB * CELLD * 4);
  float* cB     = (float*)take((size_t)BB * CELLD * 4);
  h16*   hxF0   = (h16*)  take((size_t)BB * CELLD * 2);
  h16*   hxF1   = (h16*)  take((size_t)BB * CELLD * 2);
  h16*   hxB0   = (h16*)  take((size_t)BB * CELLD * 2);
  h16*   hxB1   = (h16*)  take((size_t)BB * CELLD * 2);
  float* wsum   = (float*)take((size_t)BB * TT * 4);
  float* ctxin  = (float*)take((size_t)BB * HHD * 4);
  h16*   ctxinH = (h16*)  take((size_t)BB * HHD * 2);
  unsigned* bar = (unsigned*)take(256);

  prep_weights<<<(2 * GG * KD) / 256, 256, 0, stream>>>(WihF, WhhF, WihB, WhhB, WfH, WbH);
  prep_f32_to_f16<<<((size_t)TT * BB * II) / 256, 256, 0, stream>>>(x, xH);
  prep_f32_to_f16<<<((size_t)HHD * HHD) / 256, 256, 0, stream>>>(Wctx, WcH);
  init_misc<<<128, 256, 0, stream>>>(bihF, bhhF, bihB, bhhB, biasF, biasB,
                                     hF, cF, hB, cB, hxF0, hxF1, hxB0, hxB1,
                                     wsum, bar);

  lstm_persistent<<<NBLK, 256, 0, stream>>>(xH, WfH, WbH, biasF, biasB, len,
                                            hF, cF, hB, cB,
                                            hxF0, hxF1, hxB0, hxB1,
                                            dout, outH, bar);

  attn_kernel<<<dim3(TT / 16, BB), 256, 0, stream>>>(outH, wsum);
  ctx_gemv<<<dim3(HHD / 256, BB), 256, 0, stream>>>(wsum, dout, ctxin, ctxinH);
  ctx_proj<<<HHD / 128, 256, 0, stream>>>(ctxinH, WcH, bctx, dout + OUT_CTX);
  finalize<<<(2 * BB * HHD) / 256, 256, 0, stream>>>(hF, hB, cF, cB, dout, OUT_H, OUT_C);
}
